// DynamicalInferenceLayer_83270825935573
// MI455X (gfx1250) — matrix operand order repaired and checked by reading, not compiled, not run on hardware
//
#include <hip/hip_runtime.h>

#define B_ROWS 16384
#define DIMQ   512
#define HIDN   256
#define DT_C   0.1f
#define TRY_TDM 1

#if TRY_TDM && __has_builtin(__builtin_amdgcn_tensor_load_to_lds)
#define HAVE_TDM 1
#else
#define HAVE_TDM 0
#endif

typedef __attribute__((ext_vector_type(16))) __bf16    v16bf;
typedef __attribute__((ext_vector_type(8)))  float     v8f;
typedef __attribute__((ext_vector_type(4)))  unsigned  u32x4;
typedef __attribute__((ext_vector_type(8)))  int       i32x8;
typedef __attribute__((ext_vector_type(4)))  int       i32x4;

__device__ __forceinline__ unsigned packf2(float a, float b) {
  // two fp32 -> packed bf16x2 (truncation)
  return (__float_as_uint(a) >> 16) | (__float_as_uint(b) & 0xffff0000u);
}
__device__ __forceinline__ unsigned short f2bf(float a) {
  return (unsigned short)(__float_as_uint(a) >> 16);
}
__device__ __forceinline__ float bf2f(unsigned short h) {
  return __uint_as_float(((unsigned)h) << 16);
}

__device__ __forceinline__ float fast_tanh(float x) {
#if __has_builtin(__builtin_amdgcn_tanhf)
  return __builtin_amdgcn_tanhf(x);   // v_tanh_f32 on gfx1250
#else
  return tanhf(x);
#endif
}

__device__ __forceinline__ v8f wmma_bf16(v16bf a, v16bf b, v8f c) {
  // D = A(16x32 bf16) * B(32x16 bf16) + C(16x16 f32)
  return __builtin_amdgcn_wmma_f32_16x16x32_bf16(false, a, false, b, (short)0, c,
                                                 false, false);
}

// Issue an async stage of `dwords` DWORDs (contiguous global -> LDS).
// TDM path: Tensor Data Mover DMA (TENSORcnt), issued by wave 0 only.
__device__ __forceinline__ void stage_issue(unsigned* dst, const unsigned* src,
                                            int dwords, int tid) {
#if HAVE_TDM
  if (tid < 32) {
    unsigned lds = (unsigned)(unsigned long long)(size_t)dst;  // LDS byte offset
    unsigned long long ga = (unsigned long long)(size_t)src;
    unsigned dw = (unsigned)dwords;
    u32x4 g0;
    g0.x = 1u;                                       // count=1, is_restore=0
    g0.y = lds;                                      // lds_addr [63:32]
    g0.z = (unsigned)ga;                             // global_addr [95:64]
    g0.w = (unsigned)((ga >> 32) & 0x1ffffffull)     // global_addr [120:96]
           | (2u << 30);                             // type=2 ("image")
    i32x8 g1;
    g1[0] = (int)(2u << 16);                         // data_size=2 (4B), wg_mask=0
    g1[1] = (int)(dw << 16);                         // tensor_dim0 lo16 @bit48
    g1[2] = (int)((dw >> 16) | (1u << 16));          // tensor_dim0 hi | tensor_dim1=1
    g1[3] = (int)(dw << 16);                         // tile_dim0 @bits127:112
    g1[4] = 1;                                       // tile_dim1=1
    g1[5] = (int)dw;                                 // tensor_dim0_stride
    g1[6] = 0;
    g1[7] = 0;
    i32x4 g2 = {0, 0, 0, 0};
    i32x4 g3 = {0, 0, 0, 0};
    i32x8 g4 = {0, 0, 0, 0, 0, 0, 0, 0};
    __builtin_amdgcn_tensor_load_to_lds(g0, g1, g2, g3, g4, 0);
  }
#else
  const uint4* s = (const uint4*)src;
  uint4* d = (uint4*)dst;
  int n4 = dwords >> 2;
  for (int i = tid; i < n4; i += 128) d[i] = s[i];
#endif
}

__device__ __forceinline__ void stage_wait(int tid) {
#if HAVE_TDM
  if (tid < 32) __builtin_amdgcn_s_wait_tensorcnt(0);
#endif
}

// One A-fragment row of NT WMMAs. All NT B-fragments are preloaded into
// registers first: NT live register sets force the scheduler to queue the
// ds_load_b128s ahead of the WMMA chain (staggered s_wait_dscnt instead of
// a full s_wait_dscnt 0 before every wmma).
template <int NT>
__device__ __forceinline__ void mma_row(v8f* acc, v16bf a,
                                        const unsigned* __restrict__ wbase, int lane) {
  union Bu { v16bf v; uint4 qd[2]; } b[NT];
#pragma unroll
  for (int nt = 0; nt < NT; ++nt) {
    const uint4* bs = (const uint4*)(wbase + nt * 256 + lane * 8);
    b[nt].qd[0] = bs[0];
    b[nt].qd[1] = bs[1];
  }
#pragma unroll
  for (int nt = 0; nt < NT; ++nt)
    acc[nt] = wmma_bf16(a, b[nt].v, acc[nt]);
}

// Double-buffered, TDM-pipelined GEMM over KB k-blocks of 32.
// Stage(kb+1) is in flight while computing on stage(kb).
// NT = column tiles per k-block; NSPLIT = how many mma_row chunks (B-register cap).
template <int NT, int NSPLIT, int KB, typename F>
__device__ __forceinline__ void gemm_pipe(v8f* acc, const unsigned* __restrict__ Wsrc,
                                          int stageDwords, int srcStrideDwords,
                                          unsigned (&wbuf)[2][8192], int tid, int lane,
                                          F&& aload) {
  constexpr int NCH = NT / NSPLIT;
  stage_issue(wbuf[0], Wsrc, stageDwords, tid);
  stage_wait(tid);
  __syncthreads();
  for (int kb = 0; kb < KB; ++kb) {
    if (kb + 1 < KB) {
      stage_issue(wbuf[(kb + 1) & 1], Wsrc + (size_t)(kb + 1) * srcStrideDwords,
                  stageDwords, tid);
      __builtin_prefetch(Wsrc + (size_t)(kb + 1) * srcStrideDwords, 0, 0);
    }
    v16bf a = aload(kb);
#pragma unroll
    for (int ch = 0; ch < NSPLIT; ++ch)
      mma_row<NCH>(acc + ch * NCH, a, wbuf[kb & 1] + ch * NCH * 256, lane);
    if (kb + 1 < KB) {
      stage_wait(tid);
      __syncthreads();
    }
  }
}

// ---------------------------------------------------------------------------
// Prep kernels
// ---------------------------------------------------------------------------
__global__ void init_state_kernel(const float* __restrict__ z,
                                  float* __restrict__ q, float* __restrict__ p,
                                  int n) {
  int i = blockIdx.x * blockDim.x + threadIdx.x;
  if (i < n) { q[i] = z[i]; p[i] = 0.f; }
}

// Repack fp32 (Ktot x Ntot) matrix (optionally a transposed view of the source)
// into WMMA B-operand tile layout: per (kb, nt) 32x16 bf16 block of 256 DWORDs,
// DWORD index = lane*8 + vg, content = pair(W[k0], W[k0+1]) for column nt*16+lane%16,
// k0 = kb*32 + (lane<16 ? 0 : 16) + 2*vg.
__global__ void swizzle_kernel(const float* __restrict__ W,
                               unsigned* __restrict__ dst,
                               int Ktot, int Ntot, int transposed) {
  int total = (Ktot * Ntot) >> 1;
  int d = blockIdx.x * 256 + threadIdx.x;
  if (d >= total) return;
  int j = d & 255;
  int blockId = d >> 8;
  int ntiles = Ntot >> 4;
  int kb = blockId / ntiles;
  int nt = blockId - kb * ntiles;
  int c  = j >> 3;
  int vg = j & 7;
  int col = nt * 16 + (c & 15);
  int k0  = kb * 32 + ((c >> 4) << 4) + (vg << 1);
  float a, b;
  if (transposed) {  // logical(k,n) = W[n*Ktot + k]
    a = W[(size_t)col * Ktot + k0];
    b = W[(size_t)col * Ktot + k0 + 1];
  } else {           // logical(k,n) = W[k*Ntot + n]
    a = W[(size_t)k0 * Ntot + col];
    b = W[(size_t)(k0 + 1) * Ntot + col];
  }
  dst[d] = packf2(a, b);
}

// ---------------------------------------------------------------------------
// Fused gradH + leapfrog update.
// mode 0: p -= 0.5*dt*g_q   (backward GEMM uses W1^T cols [0,512))
// mode 1: q +=     dt*g_p   (backward GEMM uses W1^T cols [512,1024))
// ---------------------------------------------------------------------------
__global__ void __launch_bounds__(128)
hnn_step(float* __restrict__ q, float* __restrict__ p,
         const unsigned* __restrict__ W1swz, const unsigned* __restrict__ W2swz,
         const unsigned* __restrict__ W2Tswz, const unsigned* __restrict__ W1Tswz,
         const float* __restrict__ b1, const float* __restrict__ b2,
         const float* __restrict__ W3, int mode) {
  __shared__ __align__(16) unsigned       smemW[2][8192];    // 2 x 32KB weight staging
  __shared__ __align__(16) unsigned short smemA[64 * 256];   // h1 then u1 (bf16)
  __shared__ __align__(16) unsigned short smemU[64 * 256];   // u2 (bf16)

  const int tid   = threadIdx.x;
  const int lane  = tid & 31;
  const int wave  = tid >> 5;
  const int laneM = lane & 15;
  const int half  = lane >> 4;            // A/B K-half select (this code's k = 16*half + s)
  const int rowBase = blockIdx.x * 64 + wave * 16;

  const v8f vzero = {0.f, 0.f, 0.f, 0.f, 0.f, 0.f, 0.f, 0.f};
  v8f acc[16];
#pragma unroll
  for (int i = 0; i < 16; ++i) acc[i] = vzero;

  // A-fragment loader: 16x32 bf16 tile row from LDS-resident bf16 activations.
  auto loadA_lds = [&](const unsigned short* sm, int kb) -> v16bf {
    const int kkh = kb * 32 + half * 16;
    union { v16bf v; uint4 qd[2]; } au;
    const int rl = wave * 16 + laneM;
    const uint4* s = (const uint4*)(sm + rl * 256 + kkh);
    au.qd[0] = s[0];
    au.qd[1] = s[1];  // +8 bf16 elements = +16 bytes
    return au.v;
  };

  // ---- GEMM1: pre1 = [q|p] @ W1, K=1024, N=256 ------------------------------
  gemm_pipe<16, 1, 32>(acc, W1swz, 4096, 4096, smemW, tid, lane, [&](int kb) -> v16bf {
    const int kkh = kb * 32 + half * 16;
    union { v16bf v; unsigned u[8]; } au;
    const int row = rowBase + laneM;
    const float* src = (kkh < DIMQ) ? (q + (size_t)row * DIMQ + kkh)
                                    : (p + (size_t)row * DIMQ + (kkh - DIMQ));
#pragma unroll
    for (int j = 0; j < 4; ++j) au.u[j]     = packf2(src[2 * j],      src[2 * j + 1]);
#pragma unroll
    for (int j = 0; j < 4; ++j) au.u[4 + j] = packf2(src[8 + 2 * j], src[8 + 2 * j + 1]);
    return au.v;
  });
  // epilogue: h1 = tanh(pre1 + b1) -> smemA (bf16, row-major 64x256)
#pragma unroll
  for (int nt = 0; nt < 16; ++nt) {
    const int col = nt * 16 + laneM;
    const float bias = b1[col];
#pragma unroll
    for (int r = 0; r < 8; ++r) {
      const int rl = wave * 16 + half * 8 + r;
      smemA[rl * 256 + col] = f2bf(fast_tanh(acc[nt][r] + bias));
    }
    acc[nt] = vzero;
  }

  // ---- GEMM2: pre2 = h1 @ W2, K=256, N=256 ----------------------------------
  gemm_pipe<16, 1, 8>(acc, W2swz, 4096, 4096, smemW, tid, lane,
                      [&](int kb) -> v16bf { return loadA_lds(smemA, kb); });
  // epilogue: h2 = tanh(pre2 + b2); u2 = (1 - h2^2) * W3[col] -> smemU
#pragma unroll
  for (int nt = 0; nt < 16; ++nt) {
    const int col = nt * 16 + laneM;
    const float bias = b2[col];
    const float w3v = W3[col];
#pragma unroll
    for (int r = 0; r < 8; ++r) {
      const int rl = wave * 16 + half * 8 + r;
      const float t = fast_tanh(acc[nt][r] + bias);
      smemU[rl * 256 + col] = f2bf((1.f - t * t) * w3v);
    }
    acc[nt] = vzero;
  }

  // ---- GEMM3: dh1 = u2 @ W2^T, K=256, N=256 ---------------------------------
  gemm_pipe<16, 1, 8>(acc, W2Tswz, 4096, 4096, smemW, tid, lane,
                      [&](int kb) -> v16bf { return loadA_lds(smemU, kb); });
  // epilogue: u1 = (1 - h1^2) * dh1, overwriting h1 in smemA
#pragma unroll
  for (int nt = 0; nt < 16; ++nt) {
    const int col = nt * 16 + laneM;
#pragma unroll
    for (int r = 0; r < 8; ++r) {
      const int rl = wave * 16 + half * 8 + r;
      const float h1v = bf2f(smemA[rl * 256 + col]);
      smemA[rl * 256 + col] = f2bf((1.f - h1v * h1v) * acc[nt][r]);
    }
  }

  // ---- GEMM4: g_half = u1 @ W1^T[:, half], K=256, N=512 ---------------------
  v8f acc4[32];
#pragma unroll
  for (int i = 0; i < 32; ++i) acc4[i] = vzero;
  const int ntBase = (mode == 1) ? 32 : 0;  // column-tile base into W1^T (1024 cols)
  // W1^T swizzle has 64 column-tiles per k-block: k-block stride = 64*256 DWORDs,
  // we stage the 32-tile (8192-DWORD) half selected by ntBase.
  // NSPLIT=2: two 16-tile mma_row chunks keeps B-fragment pressure at 128 VGPRs.
  gemm_pipe<32, 2, 8>(acc4, W1Tswz + (size_t)ntBase * 256, 8192, 16384, smemW, tid, lane,
                      [&](int kb) -> v16bf { return loadA_lds(smemA, kb); });
  // update: mode 0 -> p -= 0.5*dt*g_q ; mode 1 -> q += dt*g_p
  float* tgt = (mode == 1) ? q : p;
  const float scale = (mode == 1) ? DT_C : (-0.5f * DT_C);
#pragma unroll
  for (int nt = 0; nt < 32; ++nt) {
    const int col = nt * 16 + laneM;
#pragma unroll
    for (int r = 0; r < 8; ++r) {
      const int gr = rowBase + half * 8 + r;
      const size_t idx = (size_t)gr * DIMQ + col;
      tgt[idx] += scale * acc4[nt][r];
    }
  }
}

// ---------------------------------------------------------------------------
extern "C" void kernel_launch(void* const* d_in, const int* in_sizes, int n_in,
                              void* d_out, int out_size, void* d_ws, size_t ws_size,
                              hipStream_t stream) {
  const float* z  = (const float*)d_in[0];
  const float* W1 = (const float*)d_in[1];
  const float* b1 = (const float*)d_in[2];
  const float* W2 = (const float*)d_in[3];
  const float* b2 = (const float*)d_in[4];
  const float* W3 = (const float*)d_in[5];
  // b3 (d_in[6]) does not affect the gradient.

  float* q = (float*)d_out;                     // q state lives in d_out
  float* p = (float*)d_ws;                      // 16384*512 fp32 = 32MB
  unsigned* W1swz  = (unsigned*)d_ws + 8388608; // after p
  unsigned* W2swz  = W1swz + 131072;            // 1024*256/2 DWORDs
  unsigned* W2Tswz = W2swz + 32768;             // 256*256/2
  unsigned* W1Tswz = W2Tswz + 32768;            // 256*1024/2

  const int n = B_ROWS * DIMQ;
  init_state_kernel<<<(n + 255) / 256, 256, 0, stream>>>(z, q, p, n);
  swizzle_kernel<<<512, 256, 0, stream>>>(W1, W1swz, 1024, 256, 0);   // W1
  swizzle_kernel<<<128, 256, 0, stream>>>(W2, W2swz, 256, 256, 0);    // W2
  swizzle_kernel<<<128, 256, 0, stream>>>(W2, W2Tswz, 256, 256, 1);   // W2^T
  swizzle_kernel<<<512, 256, 0, stream>>>(W1, W1Tswz, 256, 1024, 1);  // W1^T

  for (int s = 0; s < 3; ++s) {
    hnn_step<<<256, 128, 0, stream>>>(q, p, W1swz, W2swz, W2Tswz, W1Tswz, b1, b2, W3, 0);
    hnn_step<<<256, 128, 0, stream>>>(q, p, W1swz, W2swz, W2Tswz, W1Tswz, b1, b2, W3, 1);
    hnn_step<<<256, 128, 0, stream>>>(q, p, W1swz, W2swz, W2Tswz, W1Tswz, b1, b2, W3, 0);
  }
}
